// ExpanderSAGE_7773890805925
// MI455X (gfx1250) — matrix operand order repaired and checked by reading, not compiled, not run on hardware
//
#include <hip/hip_runtime.h>
#include <hip/hip_bf16.h>

// ---------------------------------------------------------------------------
// ExpanderSAGE on MI455X (gfx1250): 3-layer GraphSAGE with masked linears,
// BatchNorm+ReLU, log_softmax. GEMMs via v_wmma_f32_16x16x32_bf16 (wave32),
// fully unrolled ping/pong pipeline with compile-time offsets (pure
// global_load_b128 + partial s_wait_loadcnt, no register rotation).
// ---------------------------------------------------------------------------

typedef __attribute__((ext_vector_type(16))) __bf16 v16bf;
typedef __attribute__((ext_vector_type(8)))  __bf16 v8bf;
typedef __attribute__((ext_vector_type(8)))  float  v8f;

#define N_NODES 50000
#define N_EDGES 800000
#define INDIM   128
#define HID     256
#define OUTD    47
#define OUTP    48      // padded column count for layer 3 weights
#define BN_EPS  1e-5f

// ------------------------------- utility -----------------------------------

__global__ void zero_f32_kernel(float* __restrict__ p, long long n) {
    long long i = (long long)blockIdx.x * blockDim.x + threadIdx.x;
    if (i < n) p[i] = 0.0f;
}

// deg[dst] += 1 for every edge
__global__ void degree_kernel(const long long* __restrict__ ei,
                              float* __restrict__ deg, int E) {
    int e = blockIdx.x * blockDim.x + threadIdx.x;
    if (e < E) {
        long long d = ei[(long long)E + e];
        atomicAdd(&deg[d], 1.0f);
    }
}

// deg -> 1/max(deg,1) in place
__global__ void invdeg_kernel(float* __restrict__ deg, int n) {
    int i = blockIdx.x * blockDim.x + threadIdx.x;
    if (i < n) {
        float d = deg[i];
        deg[i] = 1.0f / fmaxf(d, 1.0f);
    }
}

// agg[dst, :] += h[src, :]   (one block per edge, coalesced feature atomics)
__global__ void scatter_add_kernel(const float* __restrict__ h,
                                   const long long* __restrict__ ei,
                                   float* __restrict__ agg, int F, int E) {
    int e = blockIdx.x;
    if (e >= E) return;
    long long s = ei[e];
    long long d = ei[(long long)E + e];
    const float* hs = h + s * (long long)F;
    float*       ad = agg + d * (long long)F;
    // warm the destination lines while the gather loads stream in
    __builtin_prefetch(ad + threadIdx.x, 1, 0);
    for (int f = threadIdx.x; f < F; f += blockDim.x)
        atomicAdd(&ad[f], hs[f]);
}

// out[i] = bf16(in[i])
__global__ void cvt_bf16_kernel(const float* __restrict__ in,
                                __bf16* __restrict__ out, long long n) {
    long long i = (long long)blockIdx.x * blockDim.x + threadIdx.x;
    if (i < n) out[i] = (__bf16)in[i];
}

// out[i] = bf16(in[i] * rowscale[i >> shiftF])   (agg * inv_deg)
__global__ void cvt_bf16_rowscale_kernel(const float* __restrict__ in,
                                         const float* __restrict__ rowscale,
                                         __bf16* __restrict__ out,
                                         int shiftF, long long n) {
    long long i = (long long)blockIdx.x * blockDim.x + threadIdx.x;
    if (i < n) out[i] = (__bf16)(in[i] * rowscale[i >> shiftF]);
}

// W [F_in, F_out] f32  ->  Wt [F_out_pad, F_in] bf16 (zero-padded rows)
__global__ void prep_weight_kernel(const float* __restrict__ W,
                                   __bf16* __restrict__ Wt,
                                   int shiftFin, int FinMask,
                                   int F_out, long long total) {
    long long i = (long long)blockIdx.x * blockDim.x + threadIdx.x;
    if (i < total) {
        int c = (int)(i >> shiftFin);      // output column (row of Wt)
        int k = (int)(i & FinMask);        // input feature
        Wt[i] = (c < F_out) ? (__bf16)W[(long long)k * F_out + c] : (__bf16)0.0f;
    }
}

// --------------------------- fused SAGE GEMM --------------------------------
// out[N, Fo] = aggb @ WtL^T + hb @ WtR^T + bias   (aggb pre-scaled by inv_deg)
// One wave computes NT adjacent 16x16 output tiles sharing one A fragment.
// Both NT and FIN are compile-time, so each pass fully unrolls with immediate
// load offsets; ping/pong fragment structs keep everything in SSA form.
// Fragment layouts per CDNA5 ISA 7.12.2 (wave32):
//   A 16x32 bf16 : lane half=lane>>4, row M=lane&15;
//                  elements 0..7  <-> K = half*8 + 0..7      (contiguous)
//                  elements 8..15 <-> K = 16 + half*8 + 0..7 (contiguous)
//   B 32x16 bf16 : col N=lane&15; same K mapping as A (table order,
//                  served from the transposed weight Wt[col, :])
//   C/D 16x16 f32: element r <-> M = r + 8*half, N = lane&15

template<int NT>
struct Frag {
    v8bf alo, ahi;
    v8bf blo[NT], bhi[NT];
};

template<int NT, int FIN>
__device__ __forceinline__ void load_frag(Frag<NT>& f,
                                          const __bf16* __restrict__ arow,
                                          const __bf16* __restrict__ wb,
                                          int k0) {
    f.alo = *(const v8bf*)(arow + k0);
    f.ahi = *(const v8bf*)(arow + k0 + 16);
    #pragma unroll
    for (int t = 0; t < NT; ++t) {
        const __bf16* wp = wb + t * 16 * FIN + k0;   // compile-time offsets
        f.blo[t] = *(const v8bf*)(wp);
        f.bhi[t] = *(const v8bf*)(wp + 16);
    }
}

template<int NT>
__device__ __forceinline__ void mma_frag(const Frag<NT>& f, v8f (&acc)[NT]) {
    v16bf a = __builtin_shufflevector(f.alo, f.ahi,
                0,1,2,3,4,5,6,7,8,9,10,11,12,13,14,15);
    #pragma unroll
    for (int t = 0; t < NT; ++t) {
        v16bf b = __builtin_shufflevector(f.blo[t], f.bhi[t],
                    0,1,2,3,4,5,6,7,8,9,10,11,12,13,14,15);
        acc[t] = __builtin_amdgcn_wmma_f32_16x16x32_bf16(
                     false, a, false, b, (short)0, acc[t], false, false);
    }
}

// one GEMM pass (src @ Wt^T), pipelined ping/pong, fully unrolled
template<int NT, int FIN>
__device__ __forceinline__ void gemm_pass(const __bf16* __restrict__ arow,
                                          const __bf16* __restrict__ wb,
                                          v8f (&acc)[NT]) {
    constexpr int STEPS = FIN / 32;       // 4 or 8 (even)
    Frag<NT> X, Y;
    load_frag<NT, FIN>(X, arow, wb, 0);
    #pragma unroll
    for (int k = 0; k < STEPS; k += 2) {
        if (k + 1 < STEPS) load_frag<NT, FIN>(Y, arow, wb, (k + 1) * 32);
        mma_frag<NT>(X, acc);
        if (k + 2 < STEPS) load_frag<NT, FIN>(X, arow, wb, (k + 2) * 32);
        if (k + 1 < STEPS) mma_frag<NT>(Y, acc);
    }
}

template<int NT, int FIN>
__global__ __launch_bounds__(32)
void sage_gemm_wmma_kernel(const __bf16* __restrict__ aggb,
                           const __bf16* __restrict__ hb,
                           const __bf16* __restrict__ wtl,   // [F_out_pad, FIN]
                           const __bf16* __restrict__ wtr,   // [F_out_pad, FIN]
                           const float* __restrict__ bias,
                           float* __restrict__ out,
                           int F_out) {
    const int lane = threadIdx.x;        // wave32
    const int half = lane >> 4;
    const int l15  = lane & 15;
    const int row0 = blockIdx.x << 4;    // 3125*16 == 50000, no row guard
    const int col0 = blockIdx.y * (NT * 16);

    v8f acc[NT];
    #pragma unroll
    for (int t = 0; t < NT; ++t) acc[t] = (v8f){};

    // pass 0: (agg * inv_deg) @ WtL^T
    gemm_pass<NT, FIN>(aggb + (long long)(row0 + l15) * FIN + (half << 3),
                       wtl  + (long long)(col0 + l15) * FIN + (half << 3), acc);
    // pass 1: h @ WtR^T
    gemm_pass<NT, FIN>(hb   + (long long)(row0 + l15) * FIN + (half << 3),
                       wtr  + (long long)(col0 + l15) * FIN + (half << 3), acc);

    #pragma unroll
    for (int t = 0; t < NT; ++t) {
        int ccol = col0 + t * 16 + l15;
        if (ccol < F_out) {
            float bv = bias[ccol];
            #pragma unroll
            for (int r = 0; r < 8; ++r) {
                int m = row0 + r + (half << 3);
                out[(long long)m * F_out + ccol] = acc[t][r] + bv;
            }
        }
    }
}

// ------------------------------ BatchNorm -----------------------------------

// coalesced partial column sums: block handles a chunk of rows, thread t = col t
__global__ void colstats_kernel(const float* __restrict__ hpre,
                                float* __restrict__ sum,
                                float* __restrict__ sumsq,
                                int F, int rows_per_block, int n) {
    int col = threadIdx.x;               // blockDim.x == F (256)
    int r0 = blockIdx.x * rows_per_block;
    int r1 = r0 + rows_per_block; if (r1 > n) r1 = n;
    float s = 0.0f, sq = 0.0f;
    for (int r = r0; r < r1; ++r) {
        if (r + 16 < r1)
            __builtin_prefetch(&hpre[(long long)(r + 16) * F + col], 0, 0);
        float v = hpre[(long long)r * F + col];
        s += v; sq += v * v;
    }
    atomicAdd(&sum[col], s);
    atomicAdd(&sumsq[col], sq);
}

__global__ void bnparams_kernel(const float* __restrict__ sum,
                                const float* __restrict__ sumsq,
                                const float* __restrict__ g,
                                const float* __restrict__ be,
                                float* __restrict__ scale,
                                float* __restrict__ shift,
                                int F, float invN) {
    int c = blockIdx.x * blockDim.x + threadIdx.x;
    if (c < F) {
        float mu  = sum[c] * invN;
        float var = sumsq[c] * invN - mu * mu;
        float sc  = g[c] * rsqrtf(var + BN_EPS);
        scale[c] = sc;
        shift[c] = be[c] - mu * sc;
    }
}

// h = relu(hpre*scale+shift) in both f32 (for scatter atomics) and bf16 (GEMM A)
__global__ void bnrelu_kernel(const float* __restrict__ hpre,
                              const float* __restrict__ scale,
                              const float* __restrict__ shift,
                              float* __restrict__ h,
                              __bf16* __restrict__ hb,
                              int Fmask, long long total) {
    long long i = (long long)blockIdx.x * blockDim.x + threadIdx.x;
    if (i < total) {
        int c = (int)(i & Fmask);
        float v = fmaxf(hpre[i] * scale[c] + shift[c], 0.0f);
        h[i]  = v;
        hb[i] = (__bf16)v;
    }
}

// ------------------------------ log_softmax ---------------------------------

__global__ void logsoftmax_kernel(const float* __restrict__ hpre,
                                  float* __restrict__ out, int n) {
    int row = blockIdx.x * blockDim.x + threadIdx.x;
    if (row >= n) return;
    const float* p = hpre + (long long)row * OUTD;
    float m = -3.402823466e+38f;
    #pragma unroll
    for (int j = 0; j < OUTD; ++j) m = fmaxf(m, p[j]);
    float s = 0.0f;
    #pragma unroll
    for (int j = 0; j < OUTD; ++j) s += __expf(p[j] - m);
    float ls = m + __logf(s);
    float* o = out + (long long)row * OUTD;
    #pragma unroll
    for (int j = 0; j < OUTD; ++j) o[j] = p[j] - ls;
}

// ------------------------------- driver --------------------------------------

static inline void zero_f32(float* p, long long n, hipStream_t s) {
    zero_f32_kernel<<<(unsigned)((n + 255) / 256), 256, 0, s>>>(p, n);
}

extern "C" void kernel_launch(void* const* d_in, const int* in_sizes, int n_in,
                              void* d_out, int out_size, void* d_ws, size_t ws_size,
                              hipStream_t stream) {
    const float*     x   = (const float*)d_in[0];
    const long long* ei  = (const long long*)d_in[1];   // int64 [2, E]
    const float*     w1l = (const float*)d_in[2];
    const float*     w1r = (const float*)d_in[3];
    const float*     b1  = (const float*)d_in[4];
    const float*     g1  = (const float*)d_in[5];
    const float*     be1 = (const float*)d_in[6];
    const float*     w2l = (const float*)d_in[7];
    const float*     w2r = (const float*)d_in[8];
    const float*     b2  = (const float*)d_in[9];
    const float*     g2  = (const float*)d_in[10];
    const float*     be2 = (const float*)d_in[11];
    const float*     w3l = (const float*)d_in[12];
    const float*     w3r = (const float*)d_in[13];
    const float*     b3  = (const float*)d_in[14];
    float* out = (float*)d_out;

    // ---- workspace carve-up (f32 region then bf16 region; all 16B aligned) --
    float* Wf     = (float*)d_ws;
    float* invdeg = Wf;                                      // 50048
    float* agg    = Wf + 50048;                              // N*HID
    float* hpre   = agg  + (long long)N_NODES * HID;         // N*HID
    float* hbuf   = hpre + (long long)N_NODES * HID;         // N*HID
    float* sums   = hbuf + (long long)N_NODES * HID;         // HID
    float* sumsq  = sums  + HID;
    float* scale  = sumsq + HID;
    float* shift  = scale + HID;

    __bf16* Bf   = (__bf16*)(shift + HID);
    __bf16* xb   = Bf;                                       // N*INDIM
    __bf16* hb   = xb   + (long long)N_NODES * INDIM;        // N*HID
    __bf16* aggb = hb   + (long long)N_NODES * HID;          // N*HID
    __bf16* wt1l = aggb + (long long)N_NODES * HID;          // HID*INDIM
    __bf16* wt1r = wt1l + (long long)HID * INDIM;
    __bf16* wt2l = wt1r + (long long)HID * INDIM;            // HID*HID
    __bf16* wt2r = wt2l + (long long)HID * HID;
    __bf16* wt3l = wt2r + (long long)HID * HID;              // OUTP*HID
    __bf16* wt3r = wt3l + (long long)OUTP * HID;

    const int   rowTiles = N_NODES / 16;     // 3125, exact
    const float invN     = 1.0f / (float)N_NODES;

    // ---- weight prep: transpose + convert to bf16 (pad layer 3 to 48 cols) --
    {
        long long t1 = (long long)HID * INDIM;   // [256,128]
        prep_weight_kernel<<<(unsigned)((t1 + 255) / 256), 256, 0, stream>>>(
            w1l, wt1l, 7, INDIM - 1, HID, t1);
        prep_weight_kernel<<<(unsigned)((t1 + 255) / 256), 256, 0, stream>>>(
            w1r, wt1r, 7, INDIM - 1, HID, t1);
        long long t2 = (long long)HID * HID;     // [256,256]
        prep_weight_kernel<<<(unsigned)((t2 + 255) / 256), 256, 0, stream>>>(
            w2l, wt2l, 8, HID - 1, HID, t2);
        prep_weight_kernel<<<(unsigned)((t2 + 255) / 256), 256, 0, stream>>>(
            w2r, wt2r, 8, HID - 1, HID, t2);
        long long t3 = (long long)OUTP * HID;    // [48,256]
        prep_weight_kernel<<<(unsigned)((t3 + 255) / 256), 256, 0, stream>>>(
            w3l, wt3l, 8, HID - 1, OUTD, t3);
        prep_weight_kernel<<<(unsigned)((t3 + 255) / 256), 256, 0, stream>>>(
            w3r, wt3r, 8, HID - 1, OUTD, t3);
    }

    // ---- degrees ----
    zero_f32(invdeg, N_NODES, stream);
    degree_kernel<<<(N_EDGES + 255) / 256, 256, 0, stream>>>(ei, invdeg, N_EDGES);
    invdeg_kernel<<<(N_NODES + 255) / 256, 256, 0, stream>>>(invdeg, N_NODES);

    // ---- x -> bf16 once ----
    {
        long long tot = (long long)N_NODES * INDIM;
        cvt_bf16_kernel<<<(unsigned)((tot + 255) / 256), 256, 0, stream>>>(x, xb, tot);
    }

    // ---------------- layer 1: x[N,128] -> h1[N,256] ----------------
    zero_f32(agg, (long long)N_NODES * INDIM, stream);
    scatter_add_kernel<<<N_EDGES, 128, 0, stream>>>(x, ei, agg, INDIM, N_EDGES);
    {
        long long tot = (long long)N_NODES * INDIM;
        cvt_bf16_rowscale_kernel<<<(unsigned)((tot + 255) / 256), 256, 0, stream>>>(
            agg, invdeg, aggb, 7, tot);
    }
    sage_gemm_wmma_kernel<4, INDIM><<<dim3(rowTiles, HID / 64), 32, 0, stream>>>(
        aggb, xb, wt1l, wt1r, b1, hpre, HID);
    zero_f32(sums, 2 * HID, stream);
    colstats_kernel<<<(N_NODES + 127) / 128, HID, 0, stream>>>(
        hpre, sums, sumsq, HID, 128, N_NODES);
    bnparams_kernel<<<1, HID, 0, stream>>>(sums, sumsq, g1, be1, scale, shift, HID, invN);
    {
        long long tot = (long long)N_NODES * HID;
        bnrelu_kernel<<<(unsigned)((tot + 255) / 256), 256, 0, stream>>>(
            hpre, scale, shift, hbuf, hb, HID - 1, tot);
    }

    // ---------------- layer 2: h1[N,256] -> h2[N,256] ----------------
    zero_f32(agg, (long long)N_NODES * HID, stream);
    scatter_add_kernel<<<N_EDGES, 128, 0, stream>>>(hbuf, ei, agg, HID, N_EDGES);
    {
        long long tot = (long long)N_NODES * HID;
        cvt_bf16_rowscale_kernel<<<(unsigned)((tot + 255) / 256), 256, 0, stream>>>(
            agg, invdeg, aggb, 8, tot);
    }
    sage_gemm_wmma_kernel<4, HID><<<dim3(rowTiles, HID / 64), 32, 0, stream>>>(
        aggb, hb, wt2l, wt2r, b2, hpre, HID);
    zero_f32(sums, 2 * HID, stream);
    colstats_kernel<<<(N_NODES + 127) / 128, HID, 0, stream>>>(
        hpre, sums, sumsq, HID, 128, N_NODES);
    bnparams_kernel<<<1, HID, 0, stream>>>(sums, sumsq, g2, be2, scale, shift, HID, invN);
    {
        long long tot = (long long)N_NODES * HID;
        bnrelu_kernel<<<(unsigned)((tot + 255) / 256), 256, 0, stream>>>(
            hpre, scale, shift, hbuf, hb, HID - 1, tot);   // h2 overwrites h1
    }

    // ---------------- layer 3: h2[N,256] -> logits[N,47] ----------------
    zero_f32(agg, (long long)N_NODES * HID, stream);
    scatter_add_kernel<<<N_EDGES, 128, 0, stream>>>(hbuf, ei, agg, HID, N_EDGES);
    {
        long long tot = (long long)N_NODES * HID;
        cvt_bf16_rowscale_kernel<<<(unsigned)((tot + 255) / 256), 256, 0, stream>>>(
            agg, invdeg, aggb, 8, tot);
    }
    sage_gemm_wmma_kernel<3, HID><<<dim3(rowTiles, 1), 32, 0, stream>>>(
        aggb, hb, wt3l, wt3r, b3, hpre, OUTD);

    logsoftmax_kernel<<<(N_NODES + 127) / 128, 128, 0, stream>>>(hpre, out, N_NODES);
}
